// ACTWrapper_41497974014353
// MI455X (gfx1250) — matrix operand order repaired and checked by reading, not compiled, not run on hardware
//
#include <hip/hip_runtime.h>
#include <hip/hip_bf16.h>
#include <math.h>

// ---------------------------------------------------------------------------
// ACT (Adaptive Computation Time) wrapper, MI455X / gfx1250.
// B=32, L=1024, H=512, d_ff=2048, 5 steps.
// GEMMs (99.9% of FLOPs) run as v_wmma_f32_16x16x32_bf16 (fp32 accum).
// Both FFN GEMMs fused per 64-row strip; ReLU intermediate lives in LDS only.
// ---------------------------------------------------------------------------

typedef __bf16 bf16;
typedef __attribute__((ext_vector_type(16))) __bf16 v16bf;
typedef __attribute__((ext_vector_type(8)))  float  v8f;

#define B_      32
#define L_      1024
#define H_      512
#define DFF     2048
#define M_      (B_ * L_)          // 32768 rows
#define STEPS   5
#define THRESH  0.99f

#define MTILE   64
#define FCH     64                  // d_ff chunk width held in LDS
#define TSTRIDE (FCH + 8)           // 72 elems = 144 B rows -> keeps b128 LDS reads 16B-aligned

// ---------------- workspace layout (bytes) ----------------
// s_bf16   [M][H]  bf16 : 32 MiB @ 0
// statews  [M][H]  f32  : 64 MiB @ 32Mi        (carry: transformed from prev step)
// W1T      [DFF][H] bf16:  2 MiB @ 96Mi
// W2T      [H][DFF] bf16:  2 MiB @ 98Mi
// halting  [M]     f32  : 128KiB @ 100Mi
// uw       [M]     f32  : 128KiB @ 100Mi+128Ki
#define WS_STATE   ((size_t)32u << 20)
#define WS_W1T     ((size_t)96u << 20)
#define WS_W2T     ((size_t)98u << 20)
#define WS_HALT    ((size_t)100u << 20)
#define WS_UW      (WS_HALT + (size_t)(M_ * 4))

// ---------------------------------------------------------------------------
// One-time weight transpose + fp32->bf16 convert, so WMMA B operands are
// contiguous along K per lane (2 x 16B loads per lane per tile).
// ---------------------------------------------------------------------------
__global__ void k_convert(const float* __restrict__ W1, const float* __restrict__ W2,
                          bf16* __restrict__ W1T, bf16* __restrict__ W2T) {
  int idx = blockIdx.x * blockDim.x + threadIdx.x;   // [0, DFF*H)
  if (idx < DFF * H_) {
    int n = idx / H_, k = idx - n * H_;              // W1T[n][k] = W1[k][n]
    W1T[idx] = (bf16)W1[k * DFF + n];
    int n2 = idx / DFF, f = idx - n2 * DFF;          // W2T[n][f] = W2[f][n]
    W2T[idx] = (bf16)W2[f * H_ + n2];
  }
}

// ---------------------------------------------------------------------------
// s = state + pos_signal(l) + step_signal(step), emitted as bf16 (A operand).
// Sinusoids recomputed on the fly (TRANS ops co-execute; cheaper than memory).
// ---------------------------------------------------------------------------
__global__ void k_build_s(const float* __restrict__ x, const float* __restrict__ statews,
                          bf16* __restrict__ s, int step) {
  int idx = blockIdx.x * blockDim.x + threadIdx.x;   // [0, M*H)
  int h   = idx & (H_ - 1);
  int row = idx >> 9;                                // /512
  int l   = row & (L_ - 1);
  const float log_inc = 9.210340371976184f / 255.0f; // ln(1e4)/(H/2 - 1)
  float inv_ts = __expf(-(float)(h & 255) * log_inc);
  float ps, ss;
  if (h < 256) { ps = __sinf((float)l * inv_ts); ss = __sinf((float)step * inv_ts); }
  else         { ps = __cosf((float)l * inv_ts); ss = __cosf((float)step * inv_ts); }
  float st = (step == 0) ? x[idx] : statews[idx];
  s[idx] = (bf16)(st + ps + ss);
}

// ---------------------------------------------------------------------------
// Ponder GEMV + halting state machine. One wave32 per row (16 bf16/lane),
// shfl_xor tree reduction. Accumulates n_updates/remainders in d_out tail.
// ---------------------------------------------------------------------------
__global__ void k_ponder(const bf16* __restrict__ s, const float* __restrict__ pw,
                         const float* __restrict__ pb, float* __restrict__ halting,
                         float* __restrict__ uw, float* __restrict__ nup_out,
                         float* __restrict__ rem_out, int step) {
  int wv   = threadIdx.x >> 5;
  int lane = threadIdx.x & 31;
  int row  = blockIdx.x * 8 + wv;
  const bf16*  sp = s + (size_t)row * H_ + lane * 16;
  const float* wp = pw + step * H_ + lane * 16;
  float sum = 0.f;
#pragma unroll
  for (int i = 0; i < 16; ++i) sum += (float)sp[i] * wp[i];
#pragma unroll
  for (int m = 16; m >= 1; m >>= 1) sum += __shfl_xor(sum, m, 32);
  if (lane == 0) {
    float p     = 1.f / (1.f + __expf(-(sum + pb[step])));
    float halt  = (step == 0) ? 0.f : halting[row];
    float rem   = (step == 0) ? 0.f : rem_out[row];
    float nup   = (step == 0) ? 0.f : nup_out[row];
    float still = (halt < 1.f) ? 1.f : 0.f;
    float tmp   = halt + p * still;
    float newh  = ((tmp >  THRESH) ? 1.f : 0.f) * still;
    float run2  = ((tmp <= THRESH) ? 1.f : 0.f) * still;
    halt += p * run2;
    rem  += newh * (1.f - halt);
    halt += newh * rem;
    nup  += run2 + newh;
    halting[row] = halt;
    rem_out[row] = rem;
    nup_out[row] = nup;
    uw[row]      = p * run2 + newh * rem;
  }
}

// ---------------------------------------------------------------------------
// Fused FFN: for a 64-row strip, stream d_ff in 64-wide chunks:
//   stage 1: T = relu(A(64x512) @ W1T_chunk + b1)      -> bf16 in LDS (9 KB)
//   stage 2: acc(64x512) += T(64x64) @ W2T_chunk        -> 16 v8f tiles/wave
// Epilogue: transformed = acc+b2; carry + halting blend into d_out.
// ---------------------------------------------------------------------------
__global__ __launch_bounds__(256, 1) void k_ffn(
    const bf16* __restrict__ s, const bf16* __restrict__ W1T,
    const bf16* __restrict__ W2T, const float* __restrict__ b1,
    const float* __restrict__ b2, const float* __restrict__ uw,
    float* __restrict__ statews, float* __restrict__ new_state, int step) {
  __shared__ bf16 Tlds[MTILE][TSTRIDE];

  const int tid    = threadIdx.x;
  const int wv     = tid >> 5;
  const int lane   = tid & 31;
  const int laneHi = lane >> 4;          // 0: K low half, 1: K high half (A/B layout)
  const int laneLo = lane & 15;
  const int mloc   = (wv & 3) * 16;      // wave's 16-row strip inside the 64-row tile
  const int rowg0  = blockIdx.x * MTILE + mloc;
  const int tpair  = (wv >> 2) * 2;      // stage-1: this wave owns T n-tiles {tpair, tpair+1}
  const int n0     = (wv >> 2) * 256;    // stage-2: this wave owns output cols [n0, n0+256)

  const v8f vzero = {0.f, 0.f, 0.f, 0.f, 0.f, 0.f, 0.f, 0.f};
  v8f acc[16];
#pragma unroll
  for (int i = 0; i < 16; ++i) acc[i] = vzero;

  // A-operand row pointer: lane<16 holds K {k..k+15}, lane>=16 holds K {k+16..k+31}.
  const bf16* arow = s + (size_t)(rowg0 + laneLo) * H_;

  for (int f0 = 0; f0 < DFF; f0 += FCH) {
    if (f0 + FCH < DFF)
      __builtin_prefetch(W1T + (size_t)(f0 + FCH + tpair * 16 + laneLo) * H_, 0, 1);
    __syncthreads();   // previous chunk's stage-2 reads done before overwriting Tlds

    // ---- stage 1: two 16x16 T-tiles per wave, K over H=512 ----
    v8f c1a = vzero, c1b = vzero;
#pragma unroll
    for (int k0 = 0; k0 < H_; k0 += 32) {
      union { v16bf v; uint4 q[2]; } a, b0, b1v;
      const bf16* pa = arow + k0 + laneHi * 16;
      a.q[0] = *(const uint4*)(pa);
      a.q[1] = *(const uint4*)(pa + 8);
      const bf16* pb0 = W1T + (size_t)(f0 + (tpair + 0) * 16 + laneLo) * H_ + k0 + laneHi * 16;
      b0.q[0] = *(const uint4*)(pb0);
      b0.q[1] = *(const uint4*)(pb0 + 8);
      const bf16* pb1 = W1T + (size_t)(f0 + (tpair + 1) * 16 + laneLo) * H_ + k0 + laneHi * 16;
      b1v.q[0] = *(const uint4*)(pb1);
      b1v.q[1] = *(const uint4*)(pb1 + 8);
      c1a = __builtin_amdgcn_wmma_f32_16x16x32_bf16(false, a.v, false, b0.v,  (short)0, c1a, false, false);
      c1b = __builtin_amdgcn_wmma_f32_16x16x32_bf16(false, a.v, false, b1v.v, (short)0, c1b, false, false);
    }
    // bias + ReLU, spill T to LDS in row-major bf16 (C layout: VGPR r -> M=r(+8), N=laneLo)
    {
      float ba = b1[f0 + (tpair + 0) * 16 + laneLo];
      float bb = b1[f0 + (tpair + 1) * 16 + laneLo];
#pragma unroll
      for (int r = 0; r < 8; ++r) {
        int m = mloc + laneHi * 8 + r;
        float va = c1a[r] + ba;  va = va > 0.f ? va : 0.f;
        float vb = c1b[r] + bb;  vb = vb > 0.f ? vb : 0.f;
        Tlds[m][(tpair + 0) * 16 + laneLo] = (bf16)va;
        Tlds[m][(tpair + 1) * 16 + laneLo] = (bf16)vb;
      }
    }
    __syncthreads();

    // ---- stage 2: acc += T_chunk @ W2T_chunk ----
#pragma unroll
    for (int kk = 0; kk < FCH; kk += 32) {
      union { v16bf v; uint4 q[2]; } a2;
      const bf16* pa = &Tlds[mloc + laneLo][kk + laneHi * 16];
      a2.q[0] = *(const uint4*)(pa);
      a2.q[1] = *(const uint4*)(pa + 8);
#pragma unroll
      for (int nt = 0; nt < 16; ++nt) {
        union { v16bf v; uint4 q[2]; } b;
        const bf16* pb = W2T + (size_t)(n0 + nt * 16 + laneLo) * DFF + f0 + kk + laneHi * 16;
        b.q[0] = *(const uint4*)(pb);
        b.q[1] = *(const uint4*)(pb + 8);
        acc[nt] = __builtin_amdgcn_wmma_f32_16x16x32_bf16(false, a2.v, false, b.v, (short)0, acc[nt], false, false);
      }
    }
  }

  // ---- epilogue: bias, carry state, halting blend ----
#pragma unroll
  for (int nt = 0; nt < 16; ++nt) {
    int n = n0 + nt * 16 + laneLo;
    float bias = b2[n];
#pragma unroll
    for (int r = 0; r < 8; ++r) {
      int rowg = rowg0 + laneHi * 8 + r;
      float t = acc[nt][r] + bias;
      size_t off = (size_t)rowg * H_ + n;
      float w    = uw[rowg];
      float prev = (step == 0) ? 0.f : new_state[off];
      statews[off]   = t;                          // carry for next step's s
      new_state[off] = t * w + prev * (1.f - w);   // running ACT output
    }
  }
}

// ---------------------------------------------------------------------------
extern "C" void kernel_launch(void* const* d_in, const int* in_sizes, int n_in,
                              void* d_out, int out_size, void* d_ws, size_t ws_size,
                              hipStream_t stream) {
  (void)in_sizes; (void)n_in; (void)out_size; (void)ws_size;
  const float* x  = (const float*)d_in[0];
  const float* pw = (const float*)d_in[1];   // [S, H, 1]
  const float* pb = (const float*)d_in[2];   // [S, 1]
  const float* W1 = (const float*)d_in[3];   // [H, DFF]
  const float* b1 = (const float*)d_in[4];   // [DFF]
  const float* W2 = (const float*)d_in[5];   // [DFF, H]
  const float* b2 = (const float*)d_in[6];   // [H]

  float* new_state = (float*)d_out;                      // [M, H]
  float* nup_out   = new_state + (size_t)M_ * H_;        // [M]
  float* rem_out   = nup_out + M_;                       // [M]

  char*  ws      = (char*)d_ws;
  bf16*  sbuf    = (bf16*)(ws);
  float* statews = (float*)(ws + WS_STATE);
  bf16*  W1T     = (bf16*)(ws + WS_W1T);
  bf16*  W2T     = (bf16*)(ws + WS_W2T);
  float* halting = (float*)(ws + WS_HALT);
  float* uwbuf   = (float*)(ws + WS_UW);

  k_convert<<<(DFF * H_ + 255) / 256, 256, 0, stream>>>(W1, W2, W1T, W2T);
  for (int step = 0; step < STEPS; ++step) {
    k_build_s<<<(M_ * H_) / 256, 256, 0, stream>>>(x, statews, sbuf, step);
    k_ponder<<<M_ / 8, 256, 0, stream>>>(sbuf, pw, pb, halting, uwbuf,
                                         nup_out, rem_out, step);
    k_ffn<<<M_ / MTILE, 256, 0, stream>>>(sbuf, W1T, W2T, b1, b2, uwbuf,
                                          statews, new_state, step);
  }
}
